// SelfAttention_34686155882958
// MI455X (gfx1250) — matrix operand order repaired and checked by reading, not compiled, not run on hardware
//
#include <hip/hip_runtime.h>

// ---------------------------------------------------------------------------
// Self-attention (B=4, S=2048, D=1024, H=16, DK=64) for MI455X / gfx1250.
// bf16 WMMA (v_wmma_f32_16x16x32_bf16) + CDNA5 async global->LDS staging
// (global_load_async_to_lds_b128 / s_wait_asynccnt) with double-buffered
// K-slices in the GEMMs.
// Pipeline: cvt -> GEMM(qkv) -> flash-attention -> GEMM(out).
// Workspace use: ~88 MB.
// ---------------------------------------------------------------------------

typedef __bf16 bf16;
typedef __attribute__((ext_vector_type(16))) __bf16 v16bf;
typedef __attribute__((ext_vector_type(8)))  __bf16 v8bf;
typedef __attribute__((ext_vector_type(8)))  float  v8f;

#define INFF __builtin_inff()

__device__ __forceinline__ v8f wmma_bf16(v16bf a, v16bf b, v8f c) {
  // 8 args: (neg_a, A, neg_b, B, c_mod, C, reuse_a, reuse_b)
  return __builtin_amdgcn_wmma_f32_16x16x32_bf16(false, a, false, b, (short)0, c,
                                                 false, false);
}

// CDNA5 async copy: 16B per lane, global -> LDS, tracked by ASYNCcnt.
// ISA 08_async_tensor §4: the LDS address is taken from the vdst VGPR plus offset; generic
// LDS pointers carry the LDS byte offset in bits [31:0].
__device__ __forceinline__ void async_copy16(void* lds, const void* gptr) {
  unsigned lofs = (unsigned)(unsigned long long)lds;  // addr[31:0] = LDS offset
  asm volatile("global_load_async_to_lds_b128 %0, %1, off"
               :: "v"(lofs), "v"(gptr) : "memory");
}
__device__ __forceinline__ void async_wait0() {
  asm volatile("s_wait_asynccnt 0x0" ::: "memory");
}
__device__ __forceinline__ void async_wait4() {
  asm volatile("s_wait_asynccnt 0x4" ::: "memory");
}

// A fragment (16x32 bf16) from row-major LDS tile (ISA 7.12.2 layout).
__device__ __forceinline__ v16bf load_fragA(const bf16* base, int ld) {
  int lane = threadIdx.x & 31;
  int m = lane & 15, hf = lane >> 4;
  const bf16* p = base + m * ld;
  v8bf lo = *(const v8bf*)(p + hf * 8);
  v8bf hi = *(const v8bf*)(p + 16 + hf * 8);
  return __builtin_shufflevector(lo, hi, 0, 1, 2, 3, 4, 5, 6, 7,
                                 8, 9, 10, 11, 12, 13, 14, 15);
}

// B fragment (32x16 bf16) from an LDS tile stored TRANSPOSED as [N][K].
__device__ __forceinline__ v16bf load_fragB(const bf16* base, int ld) {
  int lane = threadIdx.x & 31;
  int n = lane & 15, hf = lane >> 4;
  const bf16* p = base + n * ld + hf * 8;
  v8bf lo = *(const v8bf*)(p);
  v8bf hi = *(const v8bf*)(p + 16);
  return __builtin_shufflevector(lo, hi, 0, 1, 2, 3, 4, 5, 6, 7,
                                 8, 9, 10, 11, 12, 13, 14, 15);
}

// ---------------------------------------------------------------------------
// Elementwise conversions
// ---------------------------------------------------------------------------
__global__ void cvt_bf16_kernel(const float* __restrict__ in,
                                bf16* __restrict__ out, long n) {
  long i = (long)blockIdx.x * blockDim.x + threadIdx.x;
  if (i < n) out[i] = (bf16)in[i];
}

// W: [K][N] fp32  ->  Wt: [N][K] bf16
__global__ void transpose_cvt_kernel(const float* __restrict__ W,
                                     bf16* __restrict__ Wt, int K, int N) {
  long i = (long)blockIdx.x * blockDim.x + threadIdx.x;
  if (i < (long)K * N) {
    int n = (int)(i / K);
    int k = (int)(i % K);
    Wt[i] = (bf16)W[(long)k * N + n];
  }
}

// ---------------------------------------------------------------------------
// GEMM:  C[M][N] = A[M][K](bf16) * Bt[N][K](bf16)^T + bias[N]
// Block tile 128(M) x 128(N), 8 waves as 4(M)x2(N); wave tile 32x64
// (2 A-frags x 4 B-frags -> 8 WMMA per 32-deep K slice).
// K slices double-buffered in LDS, filled by async global->LDS copies.
// ---------------------------------------------------------------------------
template <bool OUT_BF16>
__global__ __launch_bounds__(256) void gemm_bias_kernel(
    const bf16* __restrict__ A, const bf16* __restrict__ Bt,
    const float* __restrict__ bias, void* __restrict__ Cout,
    int M, int N, int K, int ldc) {
  __shared__ alignas(16) bf16 As[2][128][32];
  __shared__ alignas(16) bf16 Bs[2][128][32];

  int t = threadIdx.x;
  int wave = t >> 5, lane = t & 31;
  int hf = lane >> 4, nl = lane & 15;
  int mw = wave >> 1;              // 0..3 : 32-row band
  int nw = wave & 1;               // 0..1 : 64-col band
  int mBase = blockIdx.y * 128;
  int nBase = blockIdx.x * 128;

  // two 16B chunks per thread per tile (128x32 = 512 chunks)
  int c0 = t, c1 = t + 256;
  int r0 = c0 >> 2, col0 = (c0 & 3) * 8;
  int r1 = c1 >> 2, col1 = (c1 & 3) * 8;

  v8f acc[2][4] = {};

  auto issue = [&](int buf, int k0) {
    async_copy16(&As[buf][r0][col0], &A[(long)(mBase + r0) * K + k0 + col0]);
    async_copy16(&As[buf][r1][col1], &A[(long)(mBase + r1) * K + k0 + col1]);
    async_copy16(&Bs[buf][r0][col0], &Bt[(long)(nBase + r0) * K + k0 + col0]);
    async_copy16(&Bs[buf][r1][col1], &Bt[(long)(nBase + r1) * K + k0 + col1]);
  };

  issue(0, 0);
  int nIter = K >> 5;
  for (int i = 0; i < nIter; ++i) {
    int buf = i & 1;
    if (i + 1 < nIter) {
      issue(buf ^ 1, (i + 1) << 5);  // overlap next slice with compute
      async_wait4();                 // slice i landed (4 newest still moving)
    } else {
      async_wait0();
    }
    __syncthreads();

    v16bf a0 = load_fragA(&As[buf][mw * 32][0], 32);
    v16bf a1 = load_fragA(&As[buf][mw * 32 + 16][0], 32);
#pragma unroll
    for (int n = 0; n < 4; ++n) {
      v16bf b = load_fragB(&Bs[buf][nw * 64 + n * 16][0], 32);
      acc[0][n] = wmma_bf16(a0, b, acc[0][n]);
      acc[1][n] = wmma_bf16(a1, b, acc[1][n]);
    }
    __syncthreads();  // everyone done with buf before it is refilled
  }

  // Epilogue: C layout lane -> (M = vgpr + 8*half, N = lane&15)
#pragma unroll
  for (int r = 0; r < 2; ++r) {
#pragma unroll
    for (int n = 0; n < 4; ++n) {
#pragma unroll
      for (int i = 0; i < 8; ++i) {
        int row = mBase + mw * 32 + r * 16 + hf * 8 + i;
        int col = nBase + nw * 64 + n * 16 + nl;
        float v = acc[r][n][i] + bias[col];
        if (OUT_BF16)
          ((bf16*)Cout)[(long)row * ldc + col] = (bf16)v;
        else
          ((float*)Cout)[(long)row * ldc + col] = v;
      }
    }
  }
}

// ---------------------------------------------------------------------------
// Flash attention, causal, per (b, h, 128-query tile). 8 waves x 16 rows.
// Q/K tiles staged via async global->LDS; V transposed manually.
// ---------------------------------------------------------------------------
__global__ __launch_bounds__(256) void flash_attn_kernel(
    const bf16* __restrict__ qkv,  // [B*S][3D], q|k|v thirds, head h at col h*64
    bf16* __restrict__ Ob,         // [B*S][D]
    int S, int D3, int Dd) {
  __shared__ alignas(16) bf16 Qs[128][64];
  __shared__ alignas(16) bf16 Ks[32][64];
  __shared__ alignas(16) bf16 Vt[64][32];   // [dk][key]
  __shared__ alignas(16) bf16 Ps[128][32];

  int t = threadIdx.x, wave = t >> 5, lane = t & 31;
  int hf = lane >> 4, nl = lane & 15;
  int rowBase = blockIdx.x * 128;
  int h = blockIdx.y;
  int b = blockIdx.z;

  const bf16* qbase = qkv + ((long)b * S) * D3 + h * 64;

  // Q tile: 128x64 = 1024 chunks of 16B -> 4 async copies per thread
#pragma unroll
  for (int j = 0; j < 4; ++j) {
    int c = t + j * 256;
    int r = c >> 3, col = (c & 7) * 8;
    async_copy16(&Qs[r][col], &qbase[(long)(rowBase + r) * D3 + col]);
  }

  v8f o[4] = {};
  float m_run[8], l_run[8];
#pragma unroll
  for (int i = 0; i < 8; ++i) { m_run[i] = -INFF; l_run[i] = 0.0f; }

  int nkt = ((rowBase + 127) >> 5) + 1;  // causal: only tiles touching triangle
  for (int kt = 0; kt < nkt; ++kt) {
    int kBase = kt * 32;
    __syncthreads();
    // K tile: 32x64 = 256 chunks -> 1 async copy per thread
    {
      int r = t >> 3, col = (t & 7) * 8;
      async_copy16(&Ks[r][col], &qbase[(long)(kBase + r) * D3 + Dd + col]);
    }
    // V tile transposed: Vt[dk][key] (scatter, stays on the sync path)
    {
      int e = t * 8;
      int r = e >> 6, c = e & 63;
      v8bf v = *(const v8bf*)&qbase[(long)(kBase + r) * D3 + 2 * Dd + c];
#pragma unroll
      for (int j = 0; j < 8; ++j) Vt[c + j][r] = v[j];
    }
    async_wait0();
    __syncthreads();

    // ---- GEMM1: scores 16x32 per wave -------------------------------------
    v8f s0 = {}, s1 = {};
#pragma unroll
    for (int kk = 0; kk < 64; kk += 32) {
      v16bf a  = load_fragA(&Qs[wave * 16][kk], 64);
      v16bf b0 = load_fragB(&Ks[0][kk], 64);
      v16bf b1 = load_fragB(&Ks[16][kk], 64);
      s0 = wmma_bf16(a, b0, s0);
      s1 = wmma_bf16(a, b1, s1);
    }

    // ---- online softmax ---------------------------------------------------
    const float scale = 0.125f;  // 1/sqrt(64)
    float alpha[8];
#pragma unroll
    for (int i = 0; i < 8; ++i) {
      int row  = rowBase + wave * 16 + hf * 8 + i;
      int key0 = kBase + nl, key1 = key0 + 16;
      float x0 = (key0 <= row) ? s0[i] * scale : -INFF;
      float x1 = (key1 <= row) ? s1[i] * scale : -INFF;
      float mx = fmaxf(x0, x1);
      mx = fmaxf(mx, __shfl_xor(mx, 1, 32));
      mx = fmaxf(mx, __shfl_xor(mx, 2, 32));
      mx = fmaxf(mx, __shfl_xor(mx, 4, 32));
      mx = fmaxf(mx, __shfl_xor(mx, 8, 32));
      float m_new = fmaxf(m_run[i], mx);
      float p0 = __expf(x0 - m_new);
      float p1 = __expf(x1 - m_new);
      float rs = p0 + p1;
      rs += __shfl_xor(rs, 1, 32);
      rs += __shfl_xor(rs, 2, 32);
      rs += __shfl_xor(rs, 4, 32);
      rs += __shfl_xor(rs, 8, 32);
      float al = __expf(m_run[i] - m_new);
      l_run[i] = l_run[i] * al + rs;
      m_run[i] = m_new;
      alpha[i] = al;
      // C-layout -> LDS so GEMM2 can regather as A fragment
      Ps[wave * 16 + hf * 8 + i][nl]      = (bf16)p0;
      Ps[wave * 16 + hf * 8 + i][nl + 16] = (bf16)p1;
    }

#pragma unroll
    for (int n = 0; n < 4; ++n)
#pragma unroll
      for (int i = 0; i < 8; ++i) o[n][i] *= alpha[i];

    // ---- GEMM2: O += P . V ------------------------------------------------
    v16bf aP = load_fragA(&Ps[wave * 16][0], 32);
#pragma unroll
    for (int n = 0; n < 4; ++n) {
      v16bf bV = load_fragB(&Vt[n * 16][0], 32);
      o[n] = wmma_bf16(aP, bV, o[n]);
    }
  }

  // ---- normalize + write --------------------------------------------------
#pragma unroll
  for (int n = 0; n < 4; ++n)
#pragma unroll
    for (int i = 0; i < 8; ++i) {
      int row = rowBase + wave * 16 + hf * 8 + i;
      float v = o[n][i] / l_run[i];
      Ob[((long)b * S + row) * Dd + h * 64 + n * 16 + nl] = (bf16)v;
    }
}

// ---------------------------------------------------------------------------
extern "C" void kernel_launch(void* const* d_in, const int* in_sizes, int n_in,
                              void* d_out, int out_size, void* d_ws, size_t ws_size,
                              hipStream_t stream) {
  const float* x     = (const float*)d_in[0];
  const float* W_in  = (const float*)d_in[1];
  const float* b_in  = (const float*)d_in[2];
  const float* W_out = (const float*)d_in[3];
  const float* b_out = (const float*)d_in[4];
  float* y = (float*)d_out;

  const int Bb = 4, S = 2048, Dd = 1024, H = 16;
  const int M = Bb * S;       // 8192
  const int D3 = 3 * Dd;      // 3072

  char* ws = (char*)d_ws;
  bf16* xb    = (bf16*)ws;  ws += (size_t)M * Dd * sizeof(bf16);
  bf16* WinT  = (bf16*)ws;  ws += (size_t)D3 * Dd * sizeof(bf16);
  bf16* WoutT = (bf16*)ws;  ws += (size_t)Dd * Dd * sizeof(bf16);
  bf16* qkv   = (bf16*)ws;  ws += (size_t)M * D3 * sizeof(bf16);
  bf16* Ob    = (bf16*)ws;

  {
    long n = (long)M * Dd;
    cvt_bf16_kernel<<<(int)((n + 255) / 256), 256, 0, stream>>>(x, xb, n);
  }
  {
    long n = (long)D3 * Dd;
    transpose_cvt_kernel<<<(int)((n + 255) / 256), 256, 0, stream>>>(W_in, WinT, Dd, D3);
  }
  {
    long n = (long)Dd * Dd;
    transpose_cvt_kernel<<<(int)((n + 255) / 256), 256, 0, stream>>>(W_out, WoutT, Dd, Dd);
  }

  gemm_bias_kernel<true><<<dim3(D3 / 128, M / 128), 256, 0, stream>>>(
      xb, WinT, b_in, (void*)qkv, M, D3, Dd, D3);

  flash_attn_kernel<<<dim3(S / 128, H, Bb), 256, 0, stream>>>(qkv, Ob, S, D3, Dd);

  gemm_bias_kernel<false><<<dim3(Dd / 128, M / 128), 256, 0, stream>>>(
      Ob, WoutT, b_out, (void*)y, M, Dd, Dd, Dd);
}
